// RawSDP_91096256348475
// MI455X (gfx1250) — matrix operand order repaired and checked by reading, not compiled, not run on hardware
//
#include <hip/hip_runtime.h>

#define SQ  2048
#define DH  128
#define NHQ 32
#define NHK 8

#define VSTRIDE 40   // Vt row stride (halves): 80B rows, 16B aligned, conflict-free
#define KSTRIDE 136  // Kt row stride (halves): 272B rows, 16B aligned, conflict-free

typedef __attribute__((ext_vector_type(16))) __bf16 v16bf;
typedef __attribute__((ext_vector_type(4)))  __bf16 v4bf;
typedef __attribute__((ext_vector_type(2)))  __bf16 v2bf;
typedef __attribute__((ext_vector_type(8)))  float  v8f;
typedef __attribute__((ext_vector_type(4)))  float  v4f;
typedef __attribute__((ext_vector_type(2)))  float  v2f;
typedef __attribute__((ext_vector_type(4)))  unsigned int v4u;

union FragB {
    v16bf v;
    v4bf  b4[4];
    v2bf  b2[8];
    v4u   u4[2];
    unsigned short us[16];
    unsigned int   u[8];
};

union Cvt2 {
    v2bf b;
    unsigned int u;
};

union Cvt8 {
    v4bf b[2];
    v4u  u;
};

__global__ __launch_bounds__(128) void fa_causal_gqa_kernel(
    const float* __restrict__ Q, const float* __restrict__ K,
    const float* __restrict__ V, float* __restrict__ Out) {
    // double-buffered bf16 tiles: Kt natural [kv][d], Vt transposed [d][kv]
    __shared__ unsigned short Kt[2][32 * KSTRIDE];
    __shared__ unsigned short Vt[2][DH * VSTRIDE];

    const int lane = threadIdx.x & 31;
    const int wave = threadIdx.x >> 5;
    const int h    = blockIdx.y;        // q head
    const int hk   = h >> 2;            // kv head (GQA 4:1)
    const int wgq0 = blockIdx.x * 64;   // workgroup q base
    const int q0   = wgq0 + wave * 16;  // wave q base
    const int qrow = q0 + (lane & 15);  // this lane's q row (stats domain)

    // 1/sqrt(128) * log2(e): scores land in log2 domain -> v_exp_f32 is 2^x
    const float sc = 0.08838834764831845f * 1.4426950408889634f;

    const float* kb = K + (size_t)hk * SQ * DH;
    const float* vb = V + (size_t)hk * SQ * DH;
    const int nblk = wgq0 / 32 + 2; // kv blocks of 32 up to wg causal limit

    // ---- staging thread roles ----
    const int krow = threadIdx.x >> 2;       // K: kv row, d chunk of 32
    const int kd0  = (threadIdx.x & 3) * 32;
    const int spr  = threadIdx.x >> 3;       // V: row pair (2pr, 2pr+1), 16 d
    const int sd0  = (threadIdx.x & 7) * 16;

    auto stage = [&](int buf, int kv0) {
        // K tile: global f32 -> LDS bf16, natural layout, b128 stores
        const float* kp = kb + (size_t)(kv0 + krow) * DH + kd0;
#pragma unroll
        for (int g = 0; g < 4; ++g) {
            v4f a = *(const v4f*)(kp + 8 * g);
            v4f b = *(const v4f*)(kp + 8 * g + 4);
            Cvt8 c;
            c.b[0] = __builtin_convertvector(a, v4bf);
            c.b[1] = __builtin_convertvector(b, v4bf);
            *(v4u*)&Kt[buf][krow * KSTRIDE + kd0 + 8 * g] = c.u;
        }
        __builtin_prefetch(kp + 32 * DH, 0, 1);
        // V tile: global f32 -> LDS bf16 transposed; row-pair packed b32 stores
        const float* vp0 = vb + (size_t)(kv0 + 2 * spr + 0) * DH + sd0;
        const float* vp1 = vb + (size_t)(kv0 + 2 * spr + 1) * DH + sd0;
#pragma unroll
        for (int i = 0; i < 16; i += 4) {
            v4f a = *(const v4f*)(vp0 + i);
            v4f b = *(const v4f*)(vp1 + i);
#pragma unroll
            for (int j = 0; j < 4; ++j) {
                Cvt2 c;
                v2f pv = {a[j], b[j]};
                c.b = __builtin_convertvector(pv, v2bf);
                *(unsigned int*)&Vt[buf][(sd0 + i + j) * VSTRIDE + 2 * spr] = c.u;
            }
        }
        __builtin_prefetch(vp0 + 32 * DH, 0, 1);
    };

    // ---- load Q as B-fragments (4 chunks of 32 along d), scaled, bf16 ----
    const float* qptr = Q + ((size_t)h * SQ + qrow) * DH;
    FragB qb[4];
#pragma unroll
    for (int c = 0; c < 4; ++c) {
        const float* p = qptr + c * 32 + ((lane < 16) ? 0 : 16);
#pragma unroll
        for (int i = 0; i < 16; i += 4) {
            v4f f = *(const v4f*)(p + i);
            f *= sc;
            qb[c].b4[i / 4] = __builtin_convertvector(f, v4bf);
        }
    }

    v8f o[8];
#pragma unroll
    for (int nt = 0; nt < 8; ++nt) o[nt] = (v8f){};
    float m = -3.0e38f, lsum = 0.0f;

    stage(0, 0); // prologue

    for (int blk = 0; blk < nblk; ++blk) {
        const int kv0 = blk * 32;
        const int cur = blk & 1;

        __syncthreads(); // staging of `cur` done; prior reads of `cur^1` done

        // stage next block into other buffer (overlaps with compute below)
        if (blk + 1 < nblk) stage(cur ^ 1, kv0 + 32);

        if (kv0 > q0 + 15) continue; // wave-uniform skip past causal limit

        // ---- S^T = K * Q^T : A-frags from LDS Kt (immediate offsets) ----
        v8f st[2];
        const unsigned short* ktb =
            &Kt[cur][(lane & 15) * KSTRIDE + ((lane < 16) ? 0 : 16)];
#pragma unroll
        for (int t = 0; t < 2; ++t) {
            st[t] = (v8f){};
            const unsigned short* tb = ktb + t * 16 * KSTRIDE;
#pragma unroll
            for (int c = 0; c < 4; ++c) {
                FragB ka;
                ka.u4[0] = *(const v4u*)(tb + c * 32);
                ka.u4[1] = *(const v4u*)(tb + c * 32 + 8);
                st[t] = __builtin_amdgcn_wmma_f32_16x16x32_bf16(
                    false, ka.v, false, qb[c].v, (short)0, st[t], false, false);
            }
        }

        // ---- causal mask (diagonal blocks only) ----
        if (kv0 + 31 > q0) {
            const int kvq = qrow - kv0 - ((lane < 16) ? 0 : 8);
#pragma unroll
            for (int t = 0; t < 2; ++t)
#pragma unroll
                for (int r = 0; r < 8; ++r)
                    if (t * 16 + r > kvq) st[t][r] = -3.0e38f;
        }

        // ---- block max (tree, then combine lane halves) ----
        float x0 = fmaxf(fmaxf(st[0][0], st[0][1]), fmaxf(st[0][2], st[0][3]));
        float x1 = fmaxf(fmaxf(st[0][4], st[0][5]), fmaxf(st[0][6], st[0][7]));
        float x2 = fmaxf(fmaxf(st[1][0], st[1][1]), fmaxf(st[1][2], st[1][3]));
        float x3 = fmaxf(fmaxf(st[1][4], st[1][5]), fmaxf(st[1][6], st[1][7]));
        float mb = fmaxf(fmaxf(x0, x1), fmaxf(x2, x3));
        mb = fmaxf(mb, __shfl_xor(mb, 16));
        const float mn = fmaxf(m, mb);

        // ---- exponentials (base 2, v_exp_f32) + P A-frag pack (tree-summed) ----
        FragB pa;
        float rsa[4] = {0.0f, 0.0f, 0.0f, 0.0f};
#pragma unroll
        for (int t = 0; t < 2; ++t)
#pragma unroll
            for (int r = 0; r < 8; r += 2) {
                float pe0 = __builtin_amdgcn_exp2f(st[t][r + 0] - mn);
                float pe1 = __builtin_amdgcn_exp2f(st[t][r + 1] - mn);
                rsa[r / 2 + 0] += pe0;
                rsa[(r / 2 + 2) & 3] += pe1;
                v2f pv = {pe0, pe1};
                pa.b2[(t * 8 + r) / 2] = __builtin_convertvector(pv, v2bf);
            }
        float rs = (rsa[0] + rsa[1]) + (rsa[2] + rsa[3]);
        rs += __shfl_xor(rs, 16);

        // ---- O rescale only if any row's max advanced (alpha!=1 somewhere) ----
        if (__ballot(mb > m) != 0ull) {
            const float alpha = __builtin_amdgcn_exp2f(m - mn);
            const int rb = (lane < 16) ? 0 : 8;
            float af[8];
#pragma unroll
            for (int r = 0; r < 8; ++r) af[r] = __shfl(alpha, rb + r);
#pragma unroll
            for (int nt = 0; nt < 8; ++nt)
#pragma unroll
                for (int r = 0; r < 8; ++r) o[nt][r] *= af[r];
            lsum = lsum * alpha + rs;
        } else {
            lsum += rs;
        }
        m = mn;

        // ---- O += P * V  (V B-frags: 2 x ds_load_b128, immediate offsets) ----
        const int kloc = (lane < 16) ? 0 : 8;
        const unsigned short* vtb = &Vt[cur][(lane & 15) * VSTRIDE + kloc];
#pragma unroll
        for (int nt = 0; nt < 8; ++nt) {
            FragB vf;
            vf.u4[0] = *(const v4u*)(vtb + nt * 16 * VSTRIDE);
            vf.u4[1] = *(const v4u*)(vtb + nt * 16 * VSTRIDE + 16);
            o[nt] = __builtin_amdgcn_wmma_f32_16x16x32_bf16(
                false, pa.v, false, vf.v, (short)0, o[nt], false, false);
        }
    }

    // ---- normalize and store (per-reg 1/l via bpermute) ----
    const float inv = 1.0f / lsum;
    const int rb = (lane < 16) ? 0 : 8;
    float invf[8];
#pragma unroll
    for (int r = 0; r < 8; ++r) invf[r] = __shfl(inv, rb + r);

    float* op = Out + ((size_t)h * SQ + q0) * DH + (lane & 15);
#pragma unroll
    for (int nt = 0; nt < 8; ++nt)
#pragma unroll
        for (int r = 0; r < 8; ++r)
            op[(size_t)(r + rb) * DH + nt * 16] = o[nt][r] * invf[r];
}

extern "C" void kernel_launch(void* const* d_in, const int* in_sizes, int n_in,
                              void* d_out, int out_size, void* d_ws, size_t ws_size,
                              hipStream_t stream) {
    (void)in_sizes; (void)n_in; (void)out_size; (void)d_ws; (void)ws_size;
    const float* Q = (const float*)d_in[0];
    const float* K = (const float*)d_in[1];
    const float* V = (const float*)d_in[2];
    float* Out = (float*)d_out;
    dim3 grid(SQ / 64, NHQ, 1);
    dim3 block(128, 1, 1);
    fa_causal_gqa_kernel<<<grid, block, 0, stream>>>(Q, K, V, Out);
}
